// DepthModule_3934190043842
// MI455X (gfx1250) — matrix operand order repaired and checked by reading, not compiled, not run on hardware
//
#include <hip/hip_runtime.h>
#include <hip/hip_bf16.h>

// ---------------------------------------------------------------------------
// DepthModule on MI455X (gfx1250): occupancy-MLP ray depth via WMMA f16.
// k_prep: transpose+convert W2/W3 to f16 once.
// k_proposal: 128 samples/ray (1 block/ray), k_secant: 8 fused iterations,
// k_final: fused value+JVP for the implicit-gradient correction.
// Weights staged into LDS via the Tensor Data Mover (tensor_load_to_lds).
// ---------------------------------------------------------------------------

typedef __attribute__((ext_vector_type(16))) _Float16 v16h;
typedef __attribute__((ext_vector_type(8)))  _Float16 v8h;
typedef __attribute__((ext_vector_type(8)))  float    v8f;

#define NRAYS   8192
#define HDIM    256
#define SNUM    128
#define NSEC_IT 8
#define WAVES   8
#define ASTR    264   // activation LDS row stride (halves): 256 + 8 pad
#define WSTR    264   // weight LDS row stride (halves): 256 + 8 pad (528 B)

#define DEPTH_LO_C 0.0f
#define DEPTH_HI_C 2.4f
#define PAD_C      0.1f
#define EPS_C      1e-6f
#define LOGIT_TAU_C 0.0f   // log(0.5/(1-0.5)) == 0

#define WBYTES   (HDIM * WSTR * 2)         // 135168 B transposed weights
#define ABYTES   (WAVES * 16 * ASTR * 2)   // 67584 B per-wave activations
#define OFF_W4H  (WBYTES + ABYTES)         // 512 B f16 W4
#define OFF_FSCR (OFF_W4H + 512)           // 512 B per-wave out scratch
#define OFF_VALS (OFF_FSCR + 512)
#define OFF_DPROP (OFF_VALS + SNUM * 4)
#define OFF_CUBE (OFF_DPROP + SNUM * 4)
#define LDS_BYTES (OFF_CUBE + 16)          // ~204.8 KB < 320 KB/WGP

#ifndef __has_builtin
#define __has_builtin(x) 0
#endif
#if __has_builtin(__builtin_amdgcn_tensor_load_to_lds) && \
    __has_builtin(__builtin_amdgcn_s_wait_tensorcnt)
#define USE_TDM 1
#else
#define USE_TDM 0
#endif

// ------------------------------ small helpers ------------------------------

__device__ __forceinline__ float softplusf(float x) {
  // branchless & overflow-safe: max(x,0) + log(1 + exp(-|x|))
  const float e = __expf(-fabsf(x));
  return fmaxf(x, 0.0f) + __logf(1.0f + e);
}
__device__ __forceinline__ float sigmoidf(float x) {
  return __builtin_amdgcn_rcpf(1.0f + __expf(-x));
}
__device__ __forceinline__ float secantf(float dl, float fl, float dh, float fh) {
  float den = fh - fl;
  den = (fabsf(den) < 1e-12f) ? 1e-12f : den;
  return -fl * (dh - dl) / den + dl;
}
__device__ __forceinline__ v16h zero_v16h() {
  v16h z;
#pragma unroll
  for (int i = 0; i < 16; ++i) z[i] = (_Float16)0.0f;
  return z;
}
__device__ __forceinline__ v8f zero_v8f() {
  v8f z;
#pragma unroll
  for (int i = 0; i < 8; ++i) z[i] = 0.0f;
  return z;
}
__device__ __forceinline__ v16h cat8(v8h lo, v8h hi) {
  return __builtin_shufflevector(lo, hi, 0,1,2,3,4,5,6,7,8,9,10,11,12,13,14,15);
}

// ------------------- weight staging: TDM (or copy fallback) ----------------
// Source: pre-transposed f16 weights wTg[n*256 + k] (built by k_prep).
// Dest: LDS wT[n*WSTR + k]  (row = 512 B data + 16 B pad = 528 B stride).

#if USE_TDM
typedef __attribute__((ext_vector_type(4))) unsigned int tdm_v4u;
typedef __attribute__((ext_vector_type(8))) int          tdm_v8i;
typedef __attribute__((ext_vector_type(4))) int          tdm_v4i;

__device__ __forceinline__ void tdm_load_wT(void* lds_dst, const _Float16* gsrc) {
  const unsigned long long la = (unsigned long long)lds_dst;
  const unsigned lds_addr = (unsigned)la;            // LDS byte offset (addr[31:0])
  const unsigned long long ga = (unsigned long long)gsrc;
  // D# group 0: count=1 | lds_addr | global_addr | type=2 ("image")
  tdm_v4u g0;
  g0[0] = 1u;
  g0[1] = lds_addr;
  g0[2] = (unsigned)(ga & 0xFFFFFFFFull);
  g0[3] = (unsigned)((ga >> 32) & 0x01FFFFFFull) | (2u << 30);
  // D# group 1: data_size=2B, pad_enable, pad every 128 DW (512 B) by 4 DW
  // (16 B) -> LDS row stride 528 B == WSTR*2. 2-D tile 256x256, stride 256.
  tdm_v8i g1;
  g1[0] = (int)((1u << 16) | (1u << 20) | (6u << 22) | (3u << 25));
  g1[1] = (int)(256u << 16);   // tensor_dim0 = 256 (bits 79:48, low half)
  g1[2] = (int)(256u << 16);   // tensor_dim0 hi=0 | tensor_dim1 = 256
  g1[3] = (int)(256u << 16);   // tensor_dim1 hi=0 | tile_dim0 = 256
  g1[4] = (int)(256u);         // tile_dim1 = 256 | tile_dim2 = 0
  g1[5] = (int)(256u);         // tensor_dim0_stride = 256 (48b, low dword)
  g1[6] = 0;
  g1[7] = 0;
  tdm_v4i gz;
  gz[0] = 0; gz[1] = 0; gz[2] = 0; gz[3] = 0;
#if defined(__clang_major__) && (__clang_major__ >= 23)
  tdm_v8i gz8;
#pragma unroll
  for (int i = 0; i < 8; ++i) gz8[i] = 0;
  __builtin_amdgcn_tensor_load_to_lds(g0, g1, gz, gz, gz8, 0);
#else
  __builtin_amdgcn_tensor_load_to_lds(g0, g1, gz, gz, 0);
#endif
}
#endif  // USE_TDM

// Caller must barrier after this (and before, if wT was being read).
__device__ __forceinline__ void load_weights(_Float16* wT, const _Float16* __restrict__ wTg,
                                             int tid) {
#if USE_TDM
  if ((tid >> 5) == 0) {               // wave 0 issues one TDM op
    tdm_load_wT((void*)wT, wTg);
    __builtin_amdgcn_s_wait_tensorcnt(0);
  }
#else
  for (int e = tid; e < HDIM * HDIM; e += 256) {
    const int n = e >> 8, k = e & 255;
    wT[n * WSTR + k] = wTg[e];
  }
#endif
}

// --------------------------- GEMM building blocks --------------------------

// Epilogue: bias + nonlinearity, store f16 activations to per-wave LDS.
// JVP: rows 0-7 (lanes<16) = value -> softplus(z); rows 8-15 (lanes>=16) =
// tangent -> t * sigmoid(z_value), z_value fetched via shfl_xor(16).
template <bool JVP>
__device__ __forceinline__ void store_tile(_Float16* actw, v8f acc, float bn,
                                           int nt, int lane) {
  const int ncol = lane & 15;
#pragma unroll
  for (int i = 0; i < 8; ++i) {
    if (!JVP) {
      const int row = i + ((lane & 16) ? 8 : 0);
      actw[row * ASTR + nt * 16 + ncol] = (_Float16)softplusf(acc[i] + bn);
    } else {
      float zval = acc[i] + bn;               // meaningful in lanes < 16
      float zpart = __shfl_xor(zval, 16, 32); // tangent lanes get value-row z
      float res; int row;
      if (lane < 16) { res = softplusf(zval);          row = i; }
      else           { res = acc[i] * sigmoidf(zpart); row = 8 + i; }
      actw[row * ASTR + nt * 16 + ncol] = (_Float16)res;
    }
  }
}

// Dense 256->256 layer: per-wave 16x256 GEMM via v_wmma_f32_16x16x32_f16.
template <bool JVP>
__device__ __forceinline__ void dense_layer(_Float16* actw, const _Float16* wT,
                                            const float* __restrict__ bias,
                                            int lane) {
  const int r   = lane & 15;
  const int khl = (lane & 16) ? 8 : 0;   // A: lanes 16-31 hold K 8-15 / 24-31
  v16h A[8];
#pragma unroll
  for (int kt = 0; kt < 8; ++kt) {
    const _Float16* ap = actw + r * ASTR + kt * 32 + khl;
    v8h lo = *(const v8h*)ap;
    v8h hi = *(const v8h*)(ap + 16);
    A[kt] = cat8(lo, hi);
  }
  const int ncol = lane & 15;
  const int khb  = (lane & 16) ? 8 : 0;  // B: lanes 16-31 hold K 8-15 / 24-31
  for (int nt = 0; nt < 16; ++nt) {
    v8f acc = zero_v8f();
#pragma unroll
    for (int kt = 0; kt < 8; ++kt) {
      const _Float16* bp = wT + (nt * 16 + ncol) * WSTR + kt * 32 + khb;
      v8h blo = *(const v8h*)bp;
      v8h bhi = *(const v8h*)(bp + 16);
      v16h B = cat8(blo, bhi);
      acc = __builtin_amdgcn_wmma_f32_16x16x32_f16(false, A[kt], false, B,
                                                   (short)0, acc, false, false);
    }
    const float bn = bias[nt * 16 + ncol];
    store_tile<JVP>(actw, acc, bn, nt, lane);
  }
}

// Input layer 3->256: single K=32 WMMA per 16-col tile (K zero-padded).
template <bool JVP>
__device__ __forceinline__ void input_layer(_Float16* actw, v16h A,
                                            const float* __restrict__ W1g,
                                            const float* __restrict__ b1g,
                                            int lane) {
  const int ncol = lane & 15;
  const bool hihalf = (lane & 16) != 0;  // their K 8-15 are all zero for K=3
  for (int nt = 0; nt < 16; ++nt) {
    v16h B = zero_v16h();
    if (!hihalf) {
      B[0] = (_Float16)W1g[0 * HDIM + nt * 16 + ncol];
      B[1] = (_Float16)W1g[1 * HDIM + nt * 16 + ncol];
      B[2] = (_Float16)W1g[2 * HDIM + nt * 16 + ncol];
    }
    v8f acc = zero_v8f();
    acc = __builtin_amdgcn_wmma_f32_16x16x32_f16(false, A, false, B,
                                                 (short)0, acc, false, false);
    store_tile<JVP>(actw, acc, b1g[nt * 16 + ncol], nt, lane);
  }
}

// Output layer 256->1 as WMMA with B having only column 0 populated (from
// f16 W4 in LDS). Column-0 results land in lanes 0 (rows 0-7) and 16 (rows
// 8-15); redistribute via per-wave LDS scratch (same-wave DS is in-order).
// Returns the pre-bias dot product for row (lane & 15) in every lane.
__device__ __forceinline__ float out_layer(const _Float16* actw,
                                           const _Float16* w4h,
                                           float* fscrw, int lane) {
  const int r    = lane & 15;
  const int khl  = (lane & 16) ? 8 : 0;
  const int ncol = lane & 15;
  const int khb  = (lane & 16) ? 8 : 0;
  v8f acc = zero_v8f();
#pragma unroll
  for (int kt = 0; kt < 8; ++kt) {
    const _Float16* ap = actw + r * ASTR + kt * 32 + khl;
    v8h lo = *(const v8h*)ap;
    v8h hi = *(const v8h*)(ap + 16);
    v16h A = cat8(lo, hi);
    v16h B = zero_v16h();
    if (ncol == 0) {
      const _Float16* wp = w4h + kt * 32 + khb;
      v8h wlo = *(const v8h*)wp;
      v8h whi = *(const v8h*)(wp + 16);
      B = cat8(wlo, whi);
    }
    acc = __builtin_amdgcn_wmma_f32_16x16x32_f16(false, A, false, B,
                                                 (short)0, acc, false, false);
  }
  if (ncol == 0) {
    const int rbase = (lane & 16) ? 8 : 0;
#pragma unroll
    for (int i = 0; i < 8; ++i) fscrw[rbase + i] = acc[i];
  }
  return fscrw[lane & 15];
}

// Full value-MLP; returns layer-4 pre-bias output for row (lane & 15).
// Caller guarantees wT is safe to overwrite on entry (barrier upstream).
__device__ __forceinline__ float mlp_value(
    v16h Ain, _Float16* actw, _Float16* wT,
    const _Float16* __restrict__ wTg2, const _Float16* __restrict__ wTg3,
    const float* __restrict__ W1, const float* __restrict__ b1,
    const float* __restrict__ b2, const float* __restrict__ b3,
    const _Float16* w4h, float* fscrw, int tid, int lane) {
  load_weights(wT, wTg2, tid);
  input_layer<false>(actw, Ain, W1, b1, lane);  // overlaps with staging
  __syncthreads();
  dense_layer<false>(actw, wT, b2, lane);
  __syncthreads();
  load_weights(wT, wTg3, tid);
  __syncthreads();
  dense_layer<false>(actw, wT, b3, lane);
  return out_layer(actw, w4h, fscrw, lane);
}

// ------------------------------- kernel 0 ----------------------------------
// One-time: W2/W3 (f32, row-major [k][n]) -> transposed f16 wTg[n*256+k].
__global__ void __launch_bounds__(256, 1)
k_prep(const float* __restrict__ W2, const float* __restrict__ W3,
       _Float16* __restrict__ wTg2, _Float16* __restrict__ wTg3) {
  const int e = blockIdx.x * 256 + threadIdx.x;  // 0 .. 131071
  const int m = e >> 16;
  const int i = e & 65535;
  const int n = i >> 8, k = i & 255;
  const float v = m ? W3[k * HDIM + n] : W2[k * HDIM + n];
  (m ? wTg3 : wTg2)[i] = (_Float16)v;
}

// ------------------------------- kernel 1 ----------------------------------
// One block per ray; 128 proposal samples = 8 waves x 16 GEMM rows.
__global__ void __launch_bounds__(256, 1)
k_proposal(const float* __restrict__ ray0g, const float* __restrict__ rdg,
           const float* __restrict__ W1, const float* __restrict__ b1,
           const float* __restrict__ b2, const float* __restrict__ b3,
           const float* __restrict__ W4, const float* __restrict__ b4,
           const _Float16* __restrict__ wTg2, const _Float16* __restrict__ wTg3,
           float* __restrict__ dlow, float* __restrict__ flow,
           float* __restrict__ dhigh, float* __restrict__ fhigh,
           float* __restrict__ maskA) {
  extern __shared__ char smem[];
  _Float16* wT   = (_Float16*)smem;
  _Float16* act0 = (_Float16*)(smem + WBYTES);
  _Float16* w4h  = (_Float16*)(smem + OFF_W4H);
  float* fscr  = (float*)(smem + OFF_FSCR);
  float* vals  = (float*)(smem + OFF_VALS);
  float* dprop = (float*)(smem + OFF_DPROP);
  float* cubeb = (float*)(smem + OFF_CUBE);

  const int ray  = blockIdx.x;
  const int tid  = threadIdx.x;
  const int lane = tid & 31;
  const int wave = tid >> 5;
  _Float16* actw = act0 + wave * 16 * ASTR;
  float* fscrw = fscr + wave * 16;

  const float r0x = ray0g[ray * 3 + 0], r0y = ray0g[ray * 3 + 1], r0z = ray0g[ray * 3 + 2];
  const float rdx = rdg[ray * 3 + 0],  rdy = rdg[ray * 3 + 1],  rdz = rdg[ray * 3 + 2];

  w4h[tid] = (_Float16)W4[tid];  // 256 threads -> full W4

  if (tid == 0) {  // unit-cube intersection (faithful port)
    const float pd = 0.5f + PAD_C * 0.5f;
    const float r0a[3] = {r0x, r0y, r0z};
    const float rda[3] = {rdx, rdy, rdz};
    int cin = 0; float d0 = 0.0f, d1 = 0.0f;
    for (int j = 0; j < 6; ++j) {
      const int ax = j % 3;
      const float plane = (j < 3) ? pd : -pd;
      float den = rda[ax];
      if (fabsf(den) < 1e-9f) den = 1e-9f;
      const float di = (plane - r0a[ax]) / den;
      const float px = r0a[0] + di * rda[0];
      const float py = r0a[1] + di * rda[1];
      const float pz = r0a[2] + di * rda[2];
      const bool inside = (fabsf(px) <= pd + EPS_C) && (fabsf(py) <= pd + EPS_C) &&
                          (fabsf(pz) <= pd + EPS_C);
      if (inside) { if (cin == 0) d0 = di; else if (cin == 1) d1 = di; ++cin; }
    }
    const bool mc = (cin == 2);
    const float a = mc ? fabsf(d0) : 0.0f;
    const float b = mc ? fabsf(d1) : 0.0f;
    cubeb[0] = fminf(a, b); cubeb[1] = fmaxf(a, b); cubeb[2] = mc ? 1.0f : 0.0f;
  }
  __syncthreads();
  const float clo = cubeb[0], chi = cubeb[1];
  const bool  mc  = cubeb[2] != 0.0f;

  const int srow = wave * 16 + (lane & 15);
  const float tt = (float)srow / (float)(SNUM - 1);
  const float dp = mc ? (clo + tt * (chi - clo))
                      : (DEPTH_LO_C + tt * (DEPTH_HI_C - DEPTH_LO_C));
  if (lane < 16) dprop[srow] = dp;

  v16h A = zero_v16h();
  if (lane < 16) {
    A[0] = (_Float16)(r0x + dp * rdx);
    A[1] = (_Float16)(r0y + dp * rdy);
    A[2] = (_Float16)(r0z + dp * rdz);
  }

  const float f4  = mlp_value(A, actw, wT, wTg2, wTg3, W1, b1, b2, b3,
                              w4h, fscrw, tid, lane);
  const float val = f4 + b4[0] - LOGIT_TAU_C;
  if (lane < 16) vals[srow] = val;
  __syncthreads();

  if (tid == 0) {  // cost = sign(v[i]*v[i+1]) * (S-i); argmin = first crossing
    float best = 1.0f; int bi = SNUM - 1;
    for (int i = 0; i < SNUM - 1; ++i) {
      const float pr = vals[i] * vals[i + 1];
      const float sg = (pr > 0.0f) ? 1.0f : ((pr < 0.0f) ? -1.0f : 0.0f);
      const float cost = sg * (float)(SNUM - i);
      if (cost < best) { best = cost; bi = i; }
    }
    const float fl = vals[bi];
    const bool mk = (best < 0.0f) && (fl < 0.0f) && (vals[0] < 0.0f);
    const int ih = (bi + 1 < SNUM) ? bi + 1 : SNUM - 1;
    dlow[ray]  = dprop[bi];
    flow[ray]  = fl;
    dhigh[ray] = dprop[ih];
    fhigh[ray] = vals[ih];
    maskA[ray] = mk ? 1.0f : 0.0f;
  }
}

// ------------------------------- kernel 2 ----------------------------------
// 128 rays/block; 8 fused secant iterations, rows = rays.
__global__ void __launch_bounds__(256, 1)
k_secant(const float* __restrict__ ray0g, const float* __restrict__ rdg,
         const float* __restrict__ W1, const float* __restrict__ b1,
         const float* __restrict__ b2, const float* __restrict__ b3,
         const float* __restrict__ W4, const float* __restrict__ b4,
         const _Float16* __restrict__ wTg2, const _Float16* __restrict__ wTg3,
         float* __restrict__ dlow, float* __restrict__ flow,
         float* __restrict__ dhigh, float* __restrict__ fhigh,
         float* __restrict__ dstar) {
  extern __shared__ char smem[];
  _Float16* wT   = (_Float16*)smem;
  _Float16* act0 = (_Float16*)(smem + WBYTES);
  _Float16* w4h  = (_Float16*)(smem + OFF_W4H);
  float* fscr = (float*)(smem + OFF_FSCR);
  const int tid = threadIdx.x, lane = tid & 31, wave = tid >> 5;
  _Float16* actw = act0 + wave * 16 * ASTR;
  float* fscrw = fscr + wave * 16;

  const int myray = blockIdx.x * 128 + wave * 16 + (lane & 15);
  const float r0x = ray0g[myray * 3 + 0], r0y = ray0g[myray * 3 + 1], r0z = ray0g[myray * 3 + 2];
  const float rdx = rdg[myray * 3 + 0],  rdy = rdg[myray * 3 + 1],  rdz = rdg[myray * 3 + 2];

  w4h[tid] = (_Float16)W4[tid];

  float dl = dlow[myray], fl = flow[myray], dh = dhigh[myray], fh = fhigh[myray];
  float dpred = secantf(dl, fl, dh, fh);
  const float b4v = b4[0];

  for (int it = 0; it < NSEC_IT; ++it) {
    v16h A = zero_v16h();
    if (lane < 16) {
      A[0] = (_Float16)(r0x + dpred * rdx);
      A[1] = (_Float16)(r0y + dpred * rdy);
      A[2] = (_Float16)(r0z + dpred * rdz);
    }
    const float f4 = mlp_value(A, actw, wT, wTg2, wTg3, W1, b1, b2, b3,
                               w4h, fscrw, tid, lane);
    const float fm = f4 + b4v - LOGIT_TAU_C;
    if (fm < 0.0f) { dl = dpred; fl = fm; } else { dh = dpred; fh = fm; }
    dpred = secantf(dl, fl, dh, fh);
    __syncthreads();  // all waves done with wT before next iter restages
  }
  if (lane < 16) dstar[myray] = dpred;
}

// ------------------------------- kernel 3 ----------------------------------
// Fused value+JVP finish: 8 rays/wave; rows 0-7 value, rows 8-15 tangent(rd).
__global__ void __launch_bounds__(256, 1)
k_final(const float* __restrict__ ray0g, const float* __restrict__ rdg,
        const float* __restrict__ W1, const float* __restrict__ b1,
        const float* __restrict__ b2, const float* __restrict__ b3,
        const float* __restrict__ W4, const float* __restrict__ b4,
        const _Float16* __restrict__ wTg2, const _Float16* __restrict__ wTg3,
        const float* __restrict__ dstar, const float* __restrict__ maskA,
        float* __restrict__ out) {
  extern __shared__ char smem[];
  _Float16* wT   = (_Float16*)smem;
  _Float16* act0 = (_Float16*)(smem + WBYTES);
  _Float16* w4h  = (_Float16*)(smem + OFF_W4H);
  float* fscr = (float*)(smem + OFF_FSCR);
  const int tid = threadIdx.x, lane = tid & 31, wave = tid >> 5;
  _Float16* actw = act0 + wave * 16 * ASTR;
  float* fscrw = fscr + wave * 16;

  const int myray = blockIdx.x * 64 + wave * 8 + (lane & 7);
  const float r0x = ray0g[myray * 3 + 0], r0y = ray0g[myray * 3 + 1], r0z = ray0g[myray * 3 + 2];
  const float rdx = rdg[myray * 3 + 0],  rdy = rdg[myray * 3 + 1],  rdz = rdg[myray * 3 + 2];
  const float ds = dstar[myray];
  const float mk = maskA[myray];

  w4h[tid] = (_Float16)W4[tid];

  v16h A = zero_v16h();
  if (lane < 16) {
    if ((lane & 15) < 8) {            // value rows: surface point
      A[0] = (_Float16)(r0x + ds * rdx);
      A[1] = (_Float16)(r0y + ds * rdy);
      A[2] = (_Float16)(r0z + ds * rdz);
    } else {                          // tangent rows: dp/dd = rd
      A[0] = (_Float16)rdx;
      A[1] = (_Float16)rdy;
      A[2] = (_Float16)rdz;
    }
  }

  load_weights(wT, wTg2, tid);
  input_layer<true>(actw, A, W1, b1, lane);
  __syncthreads();
  dense_layer<true>(actw, wT, b2, lane);
  __syncthreads();
  load_weights(wT, wTg3, tid);
  __syncthreads();
  dense_layer<true>(actw, wT, b3, lane);

  const float s4 = out_layer(actw, w4h, fscrw, lane);  // row (lane&15)
  if (lane < 8) {
    const float fv = s4 + b4[0] - LOGIT_TAU_C;         // value row
    float df = fscrw[(lane & 7) + 8];                  // tangent row, same ray
    df = (fabsf(df) < 1e-6f) ? ((df < 0.0f) ? -1e-6f : 1e-6f) : df;
    const float dres = ds - fv / df;
    out[myray] = (mk != 0.0f) ? dres : 0.0f;
  }
}

// ------------------------------- launcher ----------------------------------
extern "C" void kernel_launch(void* const* d_in, const int* in_sizes, int n_in,
                              void* d_out, int out_size, void* d_ws, size_t ws_size,
                              hipStream_t stream) {
  (void)in_sizes; (void)n_in; (void)out_size; (void)ws_size;
  const float* ray0 = (const float*)d_in[0];
  const float* rd   = (const float*)d_in[1];
  const float* W1 = (const float*)d_in[2];
  const float* b1 = (const float*)d_in[3];
  const float* W2 = (const float*)d_in[4];
  const float* b2 = (const float*)d_in[5];
  const float* W3 = (const float*)d_in[6];
  const float* b3 = (const float*)d_in[7];
  const float* W4 = (const float*)d_in[8];
  const float* b4 = (const float*)d_in[9];
  float* out = (float*)d_out;
  float* ws  = (float*)d_ws;
  float* dlow  = ws + 0 * NRAYS;
  float* flow  = ws + 1 * NRAYS;
  float* dhigh = ws + 2 * NRAYS;
  float* fhigh = ws + 3 * NRAYS;
  float* maskA = ws + 4 * NRAYS;
  float* dstar = ws + 5 * NRAYS;
  _Float16* wTg2 = (_Float16*)((char*)d_ws + 6 * NRAYS * sizeof(float));
  _Float16* wTg3 = wTg2 + HDIM * HDIM;

  k_prep<<<512, 256, 0, stream>>>(W2, W3, wTg2, wTg3);
  k_proposal<<<NRAYS, 256, LDS_BYTES, stream>>>(
      ray0, rd, W1, b1, b2, b3, W4, b4, wTg2, wTg3,
      dlow, flow, dhigh, fhigh, maskA);
  k_secant<<<NRAYS / 128, 256, LDS_BYTES, stream>>>(
      ray0, rd, W1, b1, b2, b3, W4, b4, wTg2, wTg3,
      dlow, flow, dhigh, fhigh, dstar);
  k_final<<<NRAYS / 64, 256, LDS_BYTES, stream>>>(
      ray0, rd, W1, b1, b2, b3, W4, b4, wTg2, wTg3,
      dstar, maskA, out);
}
